// SelfAttentionBlock_26542897889527
// MI455X (gfx1250) — matrix operand order repaired and checked by reading, not compiled, not run on hardware
//
#include <hip/hip_runtime.h>

// ---------- types ----------
typedef __attribute__((ext_vector_type(4)))  unsigned int u32x4;
typedef __attribute__((ext_vector_type(8)))  unsigned int u32x8;
typedef __attribute__((ext_vector_type(8)))  float        v8f;
typedef __attribute__((ext_vector_type(4)))  float        f32x4;
typedef __attribute__((ext_vector_type(16))) __bf16       v16bf;

#define MEMFENCE() asm volatile("" ::: "memory")

// ---------- helpers ----------
static __device__ __forceinline__ unsigned int f2bf(float f) {
  unsigned int u = __builtin_bit_cast(unsigned int, f);
  unsigned int r = u + 0x7fffu + ((u >> 16) & 1u);   // RNE
  return r >> 16;
}
static __device__ __forceinline__ unsigned int pack2(float lo, float hi) {
  return f2bf(lo) | (f2bf(hi) << 16);
}
static __device__ __forceinline__ v16bf mkfrag(u32x4 lo, u32x4 hi) {
  u32x8 u;
  u[0] = lo[0]; u[1] = lo[1]; u[2] = lo[2]; u[3] = lo[3];
  u[4] = hi[0]; u[5] = hi[1]; u[6] = hi[2]; u[7] = hi[3];
  return __builtin_bit_cast(v16bf, u);
}
static __device__ __forceinline__ v8f vzero8() {
  v8f z;
#pragma unroll
  for (int i = 0; i < 8; ++i) z[i] = 0.0f;
  return z;
}
static __device__ __forceinline__ v8f wmma_bf(v16bf a, v16bf b, v8f c) {
  // D = A(16x32 bf16) x B(32x16 bf16) + C(16x16 f32)
  return __builtin_amdgcn_wmma_f32_16x16x32_bf16(false, a, false, b, (short)0, c, false, false);
}

// ---------- problem constants ----------
#define BATCH 4
#define CH    256
#define HW    4096

// workspace byte offsets
#define PE_OFF   (0)                      // f32 [4096*256]           4 MB
#define WBF_OFF  (PE_OFF  + (4u << 20))   // bf16 [4][256][256]       0.5 MB (wq,wk,wv,wout)
#define XP_OFF   (WBF_OFF + (512u << 10)) // bf16 pairs [4][128][4096]x2  8 MB
#define QT_OFF   (XP_OFF  + (8u << 20))   // bf16 [4][4096][256]      8 MB  ((q+pe)/16, i-major)
#define KP_OFF   (QT_OFF  + (8u << 20))   // bf16 pairs [4][128][4096]x2  8 MB (k+pe)
#define V_OFF    (KP_OFF  + (8u << 20))   // bf16 [4][256][4096]      8 MB (v+pe)

// =====================================================================
// prep: f32 -> bf16 weight conversion (wq, wk, wv, w_out), row-major
// =====================================================================
__global__ __launch_bounds__(256) void prep_w_kernel(
    const float* wq, const float* wk, const float* wv, const float* wo,
    unsigned short* wbf) {
  int idx = blockIdx.x * 256 + threadIdx.x;            // 4*65536 total
  int wsel = idx >> 16;
  int i = idx & 65535;
  const float* src = (wsel == 0) ? wq : (wsel == 1) ? wk : (wsel == 2) ? wv : wo;
  wbf[idx] = (unsigned short)f2bf(src[i]);
}

// =====================================================================
// prep: x f32 [B][256][4096] -> channel-paired bf16 dwords [B][128][4096]
// =====================================================================
__global__ __launch_bounds__(256) void prep_x_kernel(const float* x, unsigned int* xp) {
  int dd = blockIdx.x * 256 + threadIdx.x;             // 4*128*4096 dwords
  int pos = dd & 4095;
  int c2  = (dd >> 12) & 127;
  int b   = dd >> 19;
  const float* xr = x + ((size_t)(b * CH + 2 * c2)) * HW + pos;
  xp[dd] = pack2(xr[0], xr[HW]);
}

// =====================================================================
// PE: sinusoidal table -> 2-layer SiLU MLP -> pe_flat f32 [4096*256]
// one block per 16 rows of t
// =====================================================================
__global__ __launch_bounds__(256) void pe_kernel(
    const float* w1, const float* b1, const float* w2, const float* b2, float* pe) {
  __shared__ float tab[16 * 256];
  __shared__ float hb[16 * 256];
  const int tid = threadIdx.x;
  const int t0 = blockIdx.x * 16;

  {
    int d = tid;
    float freq = powf(10000.0f, -(float)d * (1.0f / 128.0f)); // 2d/dim, dim=256
#pragma unroll
    for (int tt = 0; tt < 16; ++tt) {
      float wv_ = (float)(t0 + tt) * freq;
      tab[tt * 256 + d] = (d & 1) ? cosf(wv_) : sinf(wv_);
    }
  }
  __syncthreads();

  float a[16];
  {
    float bb = b1[tid];
#pragma unroll
    for (int tt = 0; tt < 16; ++tt) a[tt] = bb;
    const float* wr = w1 + (size_t)tid * 256;
    for (int d = 0; d < 256; ++d) {
      float wv_ = wr[d];
#pragma unroll
      for (int tt = 0; tt < 16; ++tt) a[tt] += wv_ * tab[tt * 256 + d];
    }
#pragma unroll
    for (int tt = 0; tt < 16; ++tt) {
      float z = a[tt];
      hb[tt * 256 + tid] = z / (1.0f + __expf(-z));    // SiLU
    }
  }
  __syncthreads();
  {
    float bb = b2[tid];
#pragma unroll
    for (int tt = 0; tt < 16; ++tt) a[tt] = bb;
    const float* wr = w2 + (size_t)tid * 256;
    for (int d = 0; d < 256; ++d) {
      float wv_ = wr[d];
#pragma unroll
      for (int tt = 0; tt < 16; ++tt) a[tt] += wv_ * hb[tt * 256 + d];
    }
#pragma unroll
    for (int tt = 0; tt < 16; ++tt) pe[(size_t)(t0 + tt) * 256 + tid] = a[tt];
  }
}

// =====================================================================
// QKV: bf16 WMMA GEMM  W[256x256] x X[256x4096] (+bias+pe) per batch.
// grid (64 pos-blocks, 4 batch), 128 threads = 4 waves, 16 cols/wave.
// Outputs: qT (i-major, scaled 1/16), kp (channel-paired), v (c-major)
// =====================================================================
__global__ __launch_bounds__(128, 1) void qkv_kernel(
    const unsigned int* xp_d, const unsigned int* w_d,
    const float* bq, const float* bk, const float* bv, const float* pe,
    unsigned int* qT_d, unsigned int* kp_d, unsigned short* v_h) {
  const int b = blockIdx.y, pb = blockIdx.x, tid = threadIdx.x;
  const int w = tid >> 5, lane = tid & 31;
  const int n = lane & 15, h = lane >> 4;
  const int pos = pb * 64 + w * 16 + n;

  // preload X B-fragments (K = cin = 256 -> 8 fragments), reused for q,k,v
  const unsigned int* xpb = xp_d + (size_t)b * 128 * HW;
  v16bf Xf[8];
#pragma unroll
  for (int kk = 0; kk < 8; ++kk) {
    u32x8 u;
#pragma unroll
    for (int r = 0; r < 8; ++r) u[r] = xpb[(size_t)(16 * kk + r + 8 * h) * HW + pos];
    Xf[kk] = __builtin_bit_cast(v16bf, u);
  }

#pragma unroll 1
  for (int wsel = 0; wsel < 3; ++wsel) {
    const unsigned int* wd = w_d + wsel * 32768;       // 65536 bf16 per weight
    const float* bias = (wsel == 0) ? bq : (wsel == 1) ? bk : bv;

    v8f acc[16];
#pragma unroll
    for (int t = 0; t < 16; ++t) acc[t] = vzero8();

#pragma unroll
    for (int kk = 0; kk < 8; ++kk) {
#pragma unroll
      for (int t = 0; t < 16; ++t) {
        const unsigned int* pw = wd + (size_t)(16 * t + n) * 128 + 16 * kk;
        u32x4 lo = *(const u32x4*)(pw + 8 * h);
        u32x4 hi = *(const u32x4*)(pw + 8 * h + 4);
        acc[t] = wmma_bf(mkfrag(lo, hi), Xf[kk], acc[t]);
      }
    }

    // epilogue: +bias +pe (flat reshape: pe[c*4096+pos]); write layouts
#pragma unroll
    for (int t = 0; t < 16; ++t) {
      f32x4 blo = *(const f32x4*)(bias + 16 * t + 8 * h);
      f32x4 bhi = *(const f32x4*)(bias + 16 * t + 8 * h + 4);
      float vals[8];
#pragma unroll
      for (int e = 0; e < 8; ++e) {
        int oc = 16 * t + 8 * h + e;
        float pv = pe[(size_t)oc * HW + pos];
        vals[e] = acc[t][e] + ((e < 4) ? blo[e] : bhi[e - 4]) + pv;
      }
      if (wsel == 0) {                                 // qT, fold 1/sqrt(256)
        u32x4 st;
#pragma unroll
        for (int s = 0; s < 4; ++s) st[s] = pack2(vals[2 * s] * 0.0625f, vals[2 * s + 1] * 0.0625f);
        *(u32x4*)(qT_d + (size_t)(b * HW + pos) * 128 + 8 * t + 4 * h) = st;
      } else if (wsel == 1) {                          // kp channel pairs
#pragma unroll
        for (int s = 0; s < 4; ++s)
          kp_d[(size_t)(b * 128 + 8 * t + 4 * h + s) * HW + pos] = pack2(vals[2 * s], vals[2 * s + 1]);
      } else {                                         // v c-major bf16
#pragma unroll
        for (int e = 0; e < 8; ++e)
          v_h[(size_t)(b * CH + 16 * t + 8 * h + e) * HW + pos] = (unsigned short)f2bf(vals[e]);
      }
    }
  }
}

// =====================================================================
// ATTENTION (flash-style, softmax over i) + fused output 1x1 conv.
// grid (64 j-blocks, 4 batch), 128 threads = 4 waves; wave owns 16 j cols
// and all 256 output channels -> softmax stats are wave-private.
// =====================================================================
__global__ __launch_bounds__(128, 1) void attn_kernel(
    const unsigned int* qT_d, const unsigned int* kp_d, const unsigned int* v_d,
    const unsigned int* wout_d, const float* b_out, float* out) {
  __shared__ unsigned int P_lds[16 * 64];              // 4 KB  (i-pair rows x 64 j)
  __shared__ unsigned int res_lds[128 * 64];           // 32 KB (c-pair rows x 64 j)

  const int b = blockIdx.y, jb = blockIdx.x, tid = threadIdx.x;
  const int w = tid >> 5, lane = tid & 31;
  const int n = lane & 15, h = lane >> 4;
  const int jcol = jb * 64 + w * 16 + n;

  // ---- preload K B-fragments for the wave's 16 columns (64 VGPRs) ----
  const unsigned int* kpb = kp_d + (size_t)b * 128 * HW;
  v16bf Kf[8];
#pragma unroll
  for (int kk = 0; kk < 8; ++kk) {
    u32x8 u;
#pragma unroll
    for (int r = 0; r < 8; ++r) u[r] = kpb[(size_t)(16 * kk + r + 8 * h) * HW + jcol];
    Kf[kk] = __builtin_bit_cast(v16bf, u);
  }

  v8f acc[16];
#pragma unroll
  for (int t = 0; t < 16; ++t) acc[t] = vzero8();
  float m_run = -__builtin_inff();
  float l_run = 0.0f;

  const unsigned int* qTb = qT_d + (size_t)b * HW * 128;
  const unsigned int* vb  = v_d  + (size_t)b * CH * 2048;

  for (int ib = 0; ib < HW; ib += 32) {
    // ---- scores S[32 x 16] = (q + pe)^T (k + pe) / 16 ----
    v8f s0 = vzero8(), s1 = vzero8();
#pragma unroll
    for (int kk = 0; kk < 8; ++kk) {
      const unsigned int* p0 = qTb + (size_t)(ib + n) * 128 + 16 * kk;
      const unsigned int* p1 = qTb + (size_t)(ib + 16 + n) * 128 + 16 * kk;
      u32x4 lo0 = *(const u32x4*)(p0 + 8 * h);
      u32x4 hi0 = *(const u32x4*)(p0 + 8 * h + 4);
      u32x4 lo1 = *(const u32x4*)(p1 + 8 * h);
      u32x4 hi1 = *(const u32x4*)(p1 + 8 * h + 4);
      s0 = wmma_bf(mkfrag(lo0, hi0), Kf[kk], s0);
      s1 = wmma_bf(mkfrag(lo1, hi1), Kf[kk], s1);
    }
    if (ib + 32 < HW) __builtin_prefetch(qTb + (size_t)(ib + 32 + n) * 128, 0, 0);

    // ---- online softmax over the i axis (column-wise, lane-pair n<->n+16) --
    float mx = s0[0];
#pragma unroll
    for (int e = 1; e < 8; ++e) mx = fmaxf(mx, s0[e]);
#pragma unroll
    for (int e = 0; e < 8; ++e) mx = fmaxf(mx, s1[e]);
    mx = fmaxf(mx, __shfl_xor(mx, 16, 32));
    float mn = fmaxf(m_run, mx);
    float sc = __expf(m_run - mn);
    m_run = mn;

    float p0v[8], p1v[8], ps = 0.0f;
#pragma unroll
    for (int e = 0; e < 8; ++e) { p0v[e] = __expf(s0[e] - mn); ps += p0v[e]; }
#pragma unroll
    for (int e = 0; e < 8; ++e) { p1v[e] = __expf(s1[e] - mn); ps += p1v[e]; }
    ps += __shfl_xor(ps, 16, 32);
    l_run = l_run * sc + ps;
#pragma unroll
    for (int t = 0; t < 16; ++t) acc[t] = acc[t] * sc;

    // ---- P -> LDS as bf16 i-pairs (wave-private columns) ----
#pragma unroll
    for (int s = 0; s < 4; ++s) {
      P_lds[(s + 4 * h) * 64 + w * 16 + n]     = pack2(p0v[2 * s], p0v[2 * s + 1]);
      P_lds[(s + 4 * h + 8) * 64 + w * 16 + n] = pack2(p1v[2 * s], p1v[2 * s + 1]);
    }
    MEMFENCE();                                        // same-wave DS is in-order

    u32x8 up;
#pragma unroll
    for (int r = 0; r < 8; ++r) up[r] = P_lds[(r + 8 * h) * 64 + w * 16 + n];
    v16bf Pf = __builtin_bit_cast(v16bf, up);
    MEMFENCE();

    // ---- acc[c,j] += V[c, i-tile] * P ----
#pragma unroll
    for (int t = 0; t < 16; ++t) {
      const unsigned int* pv = vb + (size_t)(16 * t + n) * 2048 + (ib >> 1);
      u32x4 lo = *(const u32x4*)(pv + 8 * h);
      u32x4 hi = *(const u32x4*)(pv + 8 * h + 4);
      acc[t] = wmma_bf(mkfrag(lo, hi), Pf, acc[t]);
    }
  }

  // ---- normalize ----
  float rl = 1.0f / l_run;
#pragma unroll
  for (int t = 0; t < 16; ++t) acc[t] = acc[t] * rl;

  // ---- res -> LDS bf16 channel pairs ----
#pragma unroll
  for (int t = 0; t < 16; ++t)
#pragma unroll
    for (int s = 0; s < 4; ++s)
      res_lds[(8 * t + s + 4 * h) * 64 + w * 16 + n] = pack2(acc[t][2 * s], acc[t][2 * s + 1]);
  MEMFENCE();

  // ---- fused output conv: out[o,j] = w_out[o,:] . res[:,j] + b_out[o] ----
  v8f oacc[16];
#pragma unroll
  for (int t = 0; t < 16; ++t) {
    f32x4 blo = *(const f32x4*)(b_out + 16 * t + 8 * h);
    f32x4 bhi = *(const f32x4*)(b_out + 16 * t + 8 * h + 4);
    v8f ini;
#pragma unroll
    for (int e = 0; e < 8; ++e) ini[e] = (e < 4) ? blo[e] : bhi[e - 4];
    oacc[t] = ini;
  }
#pragma unroll
  for (int kk = 0; kk < 8; ++kk) {
    u32x8 ur;
#pragma unroll
    for (int r = 0; r < 8; ++r) ur[r] = res_lds[(16 * kk + r + 8 * h) * 64 + w * 16 + n];
    v16bf Rf = __builtin_bit_cast(v16bf, ur);
#pragma unroll
    for (int t = 0; t < 16; ++t) {
      const unsigned int* pw = wout_d + (size_t)(16 * t + n) * 128 + 16 * kk;
      u32x4 lo = *(const u32x4*)(pw + 8 * h);
      u32x4 hi = *(const u32x4*)(pw + 8 * h + 4);
      oacc[t] = wmma_bf(mkfrag(lo, hi), Rf, oacc[t]);
    }
  }

  float* ob = out + (size_t)b * CH * HW;
#pragma unroll
  for (int t = 0; t < 16; ++t)
#pragma unroll
    for (int e = 0; e < 8; ++e)
      ob[(size_t)(16 * t + e + 8 * h) * HW + jcol] = oacc[t][e];
}

// =====================================================================
extern "C" void kernel_launch(void* const* d_in, const int* in_sizes, int n_in,
                              void* d_out, int out_size, void* d_ws, size_t ws_size,
                              hipStream_t stream) {
  const float* x      = (const float*)d_in[0];
  const float* wq     = (const float*)d_in[1];
  const float* bq     = (const float*)d_in[2];
  const float* wk     = (const float*)d_in[3];
  const float* bk     = (const float*)d_in[4];
  const float* wv     = (const float*)d_in[5];
  const float* bv     = (const float*)d_in[6];
  const float* w_out  = (const float*)d_in[7];
  const float* b_out  = (const float*)d_in[8];
  const float* mlp_w1 = (const float*)d_in[9];
  const float* mlp_b1 = (const float*)d_in[10];
  const float* mlp_w2 = (const float*)d_in[11];
  const float* mlp_b2 = (const float*)d_in[12];
  float* out = (float*)d_out;

  char* ws = (char*)d_ws;
  float*          pe   = (float*)(ws + PE_OFF);
  unsigned short* wbf  = (unsigned short*)(ws + WBF_OFF);
  unsigned int*   wbfd = (unsigned int*)(ws + WBF_OFF);
  unsigned int*   xp   = (unsigned int*)(ws + XP_OFF);
  unsigned int*   qT   = (unsigned int*)(ws + QT_OFF);
  unsigned int*   kp   = (unsigned int*)(ws + KP_OFF);
  unsigned short* v_h  = (unsigned short*)(ws + V_OFF);
  unsigned int*   v_d  = (unsigned int*)(ws + V_OFF);

  prep_w_kernel<<<1024, 256, 0, stream>>>(wq, wk, wv, w_out, wbf);
  prep_x_kernel<<<8192, 256, 0, stream>>>(x, xp);
  pe_kernel<<<256, 256, 0, stream>>>(mlp_w1, mlp_b1, mlp_w2, mlp_b2, pe);
  qkv_kernel<<<dim3(64, BATCH), 128, 0, stream>>>(xp, wbfd, bq, bk, bv, pe, qT, kp, v_h);
  attn_kernel<<<dim3(64, BATCH), 128, 0, stream>>>(qT, kp, v_d, wbfd + 3 * 32768, b_out, out);
}
